// LocalAttention_3891240370390
// MI455X (gfx1250) — matrix operand order repaired and checked by reading, not compiled, not run on hardware
//
#include <hip/hip_runtime.h>
#include <math.h>

typedef __attribute__((ext_vector_type(16))) _Float16 v16h;
typedef __attribute__((ext_vector_type(8)))  float    v8f;

#define D_HEAD   64
#define HALF_CTX 128
#define CTX      256
#define WAVE_LDS 18688   // 16*288*4 (scores) + 64 (rmbuf) + 128 (rsbuf), padded

__global__ __launch_bounds__(128) void local_attn_wmma_kernel(
    const float* __restrict__ Q, const float* __restrict__ K,
    const float* __restrict__ V, float* __restrict__ O,
    int N, int L, int H)
{
    extern __shared__ char smem_raw[];
    const int lane = threadIdx.x & 31;
    const int wave = threadIdx.x >> 5;
    const int half = lane >> 4;      // which 16-lane group
    const int l15  = lane & 15;

    const int qt = blockIdx.x * 64 + wave * 16;  // query tile start
    const int hh = blockIdx.y;                   // head
    const int nn = blockIdx.z;                   // batch

    // per-wave LDS region
    float* sS    = reinterpret_cast<float*>(smem_raw + wave * WAVE_LDS); // [16][288]
    float* rmbuf = sS + 16 * 288;   // 16 row maxes
    float* rsbuf = rmbuf + 16;      // 32 partial row sums

    const float scale = 0.125f;     // 64^-0.5

    // Row offset for sequence index j collapses to cbase + j*512 (fits in u32).
    const unsigned rstride = (unsigned)H * D_HEAD;                 // 512
    const unsigned cbase   = ((unsigned)nn * L * H + (unsigned)hh) * D_HEAD;

    // ---- Load Q tile into two A fragments (16x32 f16 each) ----
    // A layout: lane holds row (lane&15); contiguous K = 16*half + [0,16)
    const unsigned qrow  = cbase + (unsigned)(qt + l15) * rstride;
    const int      abase = 16 * half;
    v16h qa[2];
#pragma unroll
    for (int f = 0; f < 2; ++f) {
        const float* p = Q + qrow + 32 * f + abase;
        float4 x0 = *(const float4*)(p);
        float4 x1 = *(const float4*)(p + 4);
        float4 x2 = *(const float4*)(p + 8);
        float4 x3 = *(const float4*)(p + 12);
        v16h a;
        a[0]=(_Float16)x0.x; a[1]=(_Float16)x0.y; a[2]=(_Float16)x0.z; a[3]=(_Float16)x0.w;
        a[4]=(_Float16)x1.x; a[5]=(_Float16)x1.y; a[6]=(_Float16)x1.z; a[7]=(_Float16)x1.w;
        a[8]=(_Float16)x2.x; a[9]=(_Float16)x2.y; a[10]=(_Float16)x2.z; a[11]=(_Float16)x2.w;
        a[12]=(_Float16)x3.x; a[13]=(_Float16)x3.y; a[14]=(_Float16)x3.z; a[15]=(_Float16)x3.w;
        qa[f] = a;
    }

    // ---- init pad score columns [272,288) to -1e30 ----
#pragma unroll
    for (int r = 0; r < 8; ++r)
        sS[(2 * r + half) * 288 + 272 + l15] = -1e30f;

    // ---- Pass 1: scores S = Q K^T over 17 key tiles ----
    float rm[8];
#pragma unroll
    for (int g = 0; g < 8; ++g) rm[g] = -1e30f;

    for (int t = 0; t < 17; ++t) {
        const int j = qt - HALF_CTX + 16 * t + l15;    // key row this lane supplies (B layout: lane = column)
        const int jc = j < 0 ? 0 : (j >= L ? L - 1 : j);
        const unsigned krow = cbase + (unsigned)jc * rstride;

        v16h kb[2];
#pragma unroll
        for (int f = 0; f < 2; ++f) {
            // B layout: lane's 16 halves are contiguous K = 32*f + 16*half + [0,16)
            const float* p = K + krow + 32 * f + 16 * half;
            float4 b0 = *(const float4*)(p);
            float4 b1 = *(const float4*)(p + 4);
            float4 b2 = *(const float4*)(p + 8);
            float4 b3 = *(const float4*)(p + 12);
            v16h b;
            b[0]=(_Float16)b0.x; b[1]=(_Float16)b0.y; b[2]=(_Float16)b0.z; b[3]=(_Float16)b0.w;
            b[4]=(_Float16)b1.x; b[5]=(_Float16)b1.y; b[6]=(_Float16)b1.z; b[7]=(_Float16)b1.w;
            b[8]=(_Float16)b2.x; b[9]=(_Float16)b2.y; b[10]=(_Float16)b2.z; b[11]=(_Float16)b2.w;
            b[12]=(_Float16)b3.x; b[13]=(_Float16)b3.y; b[14]=(_Float16)b3.z; b[15]=(_Float16)b3.w;
            kb[f] = b;
        }

        v8f c = {};
        c = __builtin_amdgcn_wmma_f32_16x16x32_f16(false, qa[0], false, kb[0], (short)0, c, false, false);
        c = __builtin_amdgcn_wmma_f32_16x16x32_f16(false, qa[1], false, kb[1], (short)0, c, false, false);

        // C layout: element (row = g + 8*half, col = l15).
        // Branchless mask: window offset w = j-i+128 = 16t + l15 - g - 8*half (qt cancels);
        // bounds folded into one unsigned compare each -> v_cmp + v_cndmask, no exec branches.
        const bool inseq = ((unsigned)j < (unsigned)L);
#pragma unroll
        for (int g = 0; g < 8; ++g) {
            const unsigned w = (unsigned)(16 * t + l15 - g - 8 * half);
            const bool valid = (w < (unsigned)CTX) & inseq;
            const float s = valid ? c[g] * scale : -1e30f;
            rm[g] = fmaxf(rm[g], s);
            sS[(g + 8 * half) * 288 + 16 * t + l15] = s;
        }
    }

    // ---- row-max reduction within each 16-lane group, broadcast via LDS ----
#pragma unroll
    for (int off = 1; off < 16; off <<= 1)
#pragma unroll
        for (int g = 0; g < 8; ++g)
            rm[g] = fmaxf(rm[g], __shfl_xor(rm[g], off));
    if (l15 == 0) {
#pragma unroll
        for (int g = 0; g < 8; ++g) rmbuf[8 * half + g] = rm[g];
    }
    __syncthreads();
    const float mv = rmbuf[l15];   // max of this lane's A-fragment row

    // ---- Pass 2: O = softmax(S) V over 9 K-steps of 32 ----
    v8f acc[4] = {{}, {}, {}, {}};
    float ps = 0.0f;               // per-lane partial exp-sum (covers its row's columns exactly once)
    const float* srow = sS + l15 * 288;

    for (int s9 = 0; s9 < 9; ++s9) {
        const int c0 = 32 * s9 + abase;
        v16h pa;
#pragma unroll
        for (int u = 0; u < 8; ++u) {
            float e = __expf(srow[c0 + u] - mv);
            ps += e; pa[u] = (_Float16)e;
        }
#pragma unroll
        for (int u = 0; u < 8; ++u) {
            float e = __expf(srow[c0 + 8 + u] - mv);
            ps += e; pa[8 + u] = (_Float16)e;
        }

        // Hoist the 16 clamped V row offsets out of the column-tile loop.
        const int kbase = qt - HALF_CTX + 32 * s9 + 16 * half;
        unsigned voff[16];
#pragma unroll
        for (int u = 0; u < 16; ++u) {
            int j  = kbase + u;
            int jc = j < 0 ? 0 : (j >= L ? L - 1 : j);
            voff[u] = cbase + (unsigned)jc * rstride + (unsigned)l15;
        }

#pragma unroll
        for (int n0 = 0; n0 < 4; ++n0) {
            v16h vb;
#pragma unroll
            for (int u = 0; u < 16; ++u)
                vb[u] = (_Float16)V[voff[u] + (unsigned)(n0 * 16)];
            acc[n0] = __builtin_amdgcn_wmma_f32_16x16x32_f16(false, pa, false, vb, (short)0, acc[n0], false, false);
        }
    }

    // ---- row sums via LDS exchange, normalize, store ----
    rsbuf[lane] = ps;
    __syncthreads();
#pragma unroll
    for (int g = 0; g < 8; ++g) {
        const int row = g + 8 * half;
        const float inv = 1.0f / (rsbuf[row] + rsbuf[row + 16]);
        const unsigned orow = cbase + (unsigned)(qt + row) * rstride;
#pragma unroll
        for (int n0 = 0; n0 < 4; ++n0)
            O[orow + n0 * 16 + l15] = acc[n0][g] * inv;
    }
}

extern "C" void kernel_launch(void* const* d_in, const int* in_sizes, int n_in,
                              void* d_out, int out_size, void* d_ws, size_t ws_size,
                              hipStream_t stream) {
    (void)in_sizes; (void)n_in; (void)out_size; (void)d_ws; (void)ws_size;
    const float* q = (const float*)d_in[0];
    const float* k = (const float*)d_in[1];
    const float* v = (const float*)d_in[2];
    float* o = (float*)d_out;

    const int N = 4, L = 4096, H = 8;
    dim3 grid(L / 64, H, N);   // 64 x 8 x 4 workgroups
    dim3 block(128);           // 4 waves; one 16-query tile each
    size_t shmem = 4 * (size_t)WAVE_LDS;
    local_attn_wmma_kernel<<<grid, block, shmem, stream>>>(q, k, v, o, N, L, H);
}
